// MHPFGT_77300821393980
// MI455X (gfx1250) — matrix operand order repaired and checked by reading, not compiled, not run on hardware
//
#include <hip/hip_runtime.h>
#include <hip/hip_bf16.h>
#include <math.h>

// ---------------- problem constants (match reference) ----------------
#define NN      50000
#define EE      400000
#define F_IN    128
#define HID     64
#define HEADS   4
#define HEADC   16
#define NC      10
#define HV      (HEADS * NC)    // 40
#define MSZ     (HEADS * HEADC * NC) // 640 floats of M per node
#define CSTF    1e-5f

typedef __attribute__((ext_vector_type(16))) _Float16 v16h;
typedef __attribute__((ext_vector_type(8)))  float    v8f;

union AF { v16h v; _Float16 h[16]; };
union CF { v8f  v; float    f[8];  };

// A-fragment K index for 16-bit 16x32 A tile (contiguous, matches B):
// lane<16 covers K {0..15}; lane>=16 covers K {16..31}
__device__ __forceinline__ int a_kidx(int e, int half) {
    return e + 16 * half;
}

// =====================================================================
// Kernel 1: fused node transform.
//  x = relu(feat @ W_in^T + b_in)            (WMMA f16->f32, K=128)
//  Q = 1+elu(x @ W_q^T), K0 = 1+elu(x @ W_k^T), V0 = x @ W_v^T + b_v
//  hidden = hopwise[0] * V0
// One wave handles a 16-node tile; x tile staged in LDS as f16.
// =====================================================================
__global__ __launch_bounds__(256) void k_node_transform(
    const float* __restrict__ feat,
    const float* __restrict__ Win, const float* __restrict__ bin,
    const float* __restrict__ Wq,  const float* __restrict__ bq,
    const float* __restrict__ Wk,  const float* __restrict__ bk,
    const float* __restrict__ Wv,  const float* __restrict__ bv,
    const float* __restrict__ hopwise,
    float* __restrict__ Q, float* __restrict__ K0,
    float* __restrict__ V0, float* __restrict__ hidden)
{
    __shared__ _Float16 xsh[8][16][HID];   // per-wave 16x64 f16 x-tile (16 KB)

    const int lane = threadIdx.x & 31;
    const int wv   = threadIdx.x >> 5;
    const int half = lane >> 4;
    const int ml   = lane & 15;
    const int gm0  = (blockIdx.x * 8 + wv) * 16;   // N % 16 == 0
    const bool valid = (gm0 < NN);

    if (valid) {
        // ---- stage 1: x = relu(feat @ Win^T + b) via 4 K-chunks of 32 ----
        AF a[4];
        #pragma unroll
        for (int kc = 0; kc < 4; ++kc) {
            const float* frow = feat + (size_t)(gm0 + ml) * F_IN + kc * 32;
            #pragma unroll
            for (int e = 0; e < 16; ++e)
                a[kc].h[e] = (_Float16)frow[a_kidx(e, half)];
        }
        #pragma unroll
        for (int nt = 0; nt < 4; ++nt) {
            CF acc;
            #pragma unroll
            for (int r = 0; r < 8; ++r) acc.f[r] = 0.f;
            #pragma unroll
            for (int kc = 0; kc < 4; ++kc) {
                AF b;
                const float* wrow = Win + (size_t)(nt * 16 + ml) * F_IN + kc * 32 + 16 * half;
                #pragma unroll
                for (int e = 0; e < 16; ++e) b.h[e] = (_Float16)wrow[e];
                acc.v = __builtin_amdgcn_wmma_f32_16x16x32_f16(
                    false, a[kc].v, false, b.v, (short)0, acc.v, false, false);
            }
            const float bb = bin[nt * 16 + ml];
            #pragma unroll
            for (int r = 0; r < 8; ++r) {
                float xv = acc.f[r] + bb;
                xv = xv > 0.f ? xv : 0.f;
                xsh[wv][r + 8 * half][nt * 16 + ml] = (_Float16)xv;
            }
        }
    }
    __syncthreads();
    if (!valid) return;

    // ---- stage 2: A-fragments of x from LDS (2 K-chunks of 32) ----
    AF a2[2];
    #pragma unroll
    for (int kc = 0; kc < 2; ++kc)
        #pragma unroll
        for (int e = 0; e < 16; ++e)
            a2[kc].h[e] = xsh[wv][ml][kc * 32 + a_kidx(e, half)];

    // Q and K0: out dim 64 (4 n-tiles), transform 1+elu(z)
    #pragma unroll
    for (int m = 0; m < 2; ++m) {
        const float* W  = m ? Wk : Wq;
        const float* bp = m ? bk : bq;
        float*       O  = m ? K0 : Q;
        #pragma unroll
        for (int nt = 0; nt < 4; ++nt) {
            CF acc;
            #pragma unroll
            for (int r = 0; r < 8; ++r) acc.f[r] = 0.f;
            #pragma unroll
            for (int kc = 0; kc < 2; ++kc) {
                AF b;
                const float* wrow = W + (size_t)(nt * 16 + ml) * HID + kc * 32 + 16 * half;
                #pragma unroll
                for (int e = 0; e < 16; ++e) b.h[e] = (_Float16)wrow[e];
                acc.v = __builtin_amdgcn_wmma_f32_16x16x32_f16(
                    false, a2[kc].v, false, b.v, (short)0, acc.v, false, false);
            }
            const int col = nt * 16 + ml;
            const float bc = bp[col];
            #pragma unroll
            for (int r = 0; r < 8; ++r) {
                float z = acc.f[r] + bc;
                float t = z > 0.f ? 1.f + z : __expf(z);   // 1 + elu(z)
                O[(size_t)(gm0 + r + 8 * half) * HID + col] = t;
            }
        }
    }

    // V0: out dim 40 (3 n-tiles, guard col<40); hidden = hopwise[0]*V0
    const float hop0 = hopwise[0];
    #pragma unroll
    for (int nt = 0; nt < 3; ++nt) {
        const int col = nt * 16 + ml;
        CF acc;
        #pragma unroll
        for (int r = 0; r < 8; ++r) acc.f[r] = 0.f;
        #pragma unroll
        for (int kc = 0; kc < 2; ++kc) {
            AF b;
            #pragma unroll
            for (int e = 0; e < 16; ++e)
                b.h[e] = (col < HV)
                    ? (_Float16)Wv[(size_t)col * HID + kc * 32 + 16 * half + e]
                    : (_Float16)0.f;
            acc.v = __builtin_amdgcn_wmma_f32_16x16x32_f16(
                false, a2[kc].v, false, b.v, (short)0, acc.v, false, false);
        }
        if (col < HV) {
            const float bc = bv[col];
            #pragma unroll
            for (int r = 0; r < 8; ++r) {
                float z = acc.f[r] + bc;
                size_t idx = (size_t)(gm0 + r + 8 * half) * HV + col;
                V0[idx] = z;
                hidden[idx] = hop0 * z;
            }
        }
    }
}

// =====================================================================
// Kernel 2: hop-1 edge pass. One wave per edge (grid-stride):
//  dot_h = Q[d,h] . K0[s,h]  (shuffle reduction)
//  H[d]    += dot_h * V0[s]        (40 atomics)
//  Kagg[d] += K0[s]                (64 atomics)
//  M1[d]   += outer(K0[s], V0[s])  (640 atomics; M1 is L2-resident)
// =====================================================================
__global__ __launch_bounds__(256) void k_edge_hop1(
    const int* __restrict__ ei,
    const float* __restrict__ Q, const float* __restrict__ K0,
    const float* __restrict__ V0,
    float* __restrict__ M1, float* __restrict__ Kagg, float* __restrict__ H)
{
    const int lane = threadIdx.x & 31;
    const int wid  = (blockIdx.x * blockDim.x + threadIdx.x) >> 5;
    const int nw   = (gridDim.x * blockDim.x) >> 5;

    for (int e = wid; e < EE; e += nw) {
        const int s = ei[e];
        const int d = ei[EE + e];
        const float* ks = K0 + (size_t)s * HID;
        const float* vs = V0 + (size_t)s * HV;
        const float* qd = Q  + (size_t)d * HID;

        const float k_lo = ks[lane];
        const float k_hi = ks[32 + lane];

        // per-head dots: lanes 0-15/16-31 hold heads {0,2}/{1,3} partials
        float pa = qd[lane] * k_lo;
        float pb = qd[32 + lane] * k_hi;
        #pragma unroll
        for (int off = 1; off < 16; off <<= 1) {
            pa += __shfl_xor(pa, off, 32);
            pb += __shfl_xor(pb, off, 32);
        }
        const float pa2 = __shfl_xor(pa, 16, 32);
        const float pb2 = __shfl_xor(pb, 16, 32);
        float dot[4];
        if (lane < 16) { dot[0] = pa;  dot[1] = pa2; dot[2] = pb;  dot[3] = pb2; }
        else           { dot[0] = pa2; dot[1] = pa;  dot[2] = pb2; dot[3] = pb;  }

        // K aggregate
        atomicAdd(Kagg + (size_t)d * HID + lane,      k_lo);
        atomicAdd(Kagg + (size_t)d * HID + 32 + lane, k_hi);

        // H accumulate: 40 components
        {
            const int c = lane;                // 0..31
            atomicAdd(H + (size_t)d * HV + c, dot[c / 10] * vs[c]);
            if (lane < 8) {
                const int c2 = 32 + lane;      // 32..39
                atomicAdd(H + (size_t)d * HV + c2, dot[c2 / 10] * vs[c2]);
            }
        }

        // M1 += outer(K0[s], V0[s]) : 640 components, 20 per lane
        float* md = M1 + (size_t)d * MSZ;
        #pragma unroll
        for (int t = 0; t < 20; ++t) {
            const int c = lane + 32 * t;
            const int h = c / 160;
            const int r = c - h * 160;
            const int i = r / 10;
            const int j = r - i * 10;
            atomicAdd(md + c, ks[h * 16 + i] * vs[h * 10 + j]);
        }
    }
}

// =====================================================================
// Kernel 3: hop-2 edge pass. One wave per edge:
//  H[d,h,j]  += sum_i Q[d,h,i] * M1[s,h,i,j]   (reads 2.56KB of M1, L2-hot)
//  Kagg2[d]  += Kagg1[s]
// =====================================================================
__global__ __launch_bounds__(256) void k_edge_hop2(
    const int* __restrict__ ei,
    const float* __restrict__ Q, const float* __restrict__ Kagg1,
    const float* __restrict__ M1,
    float* __restrict__ Kagg2, float* __restrict__ H)
{
    const int lane = threadIdx.x & 31;
    const int wid  = (blockIdx.x * blockDim.x + threadIdx.x) >> 5;
    const int nw   = (gridDim.x * blockDim.x) >> 5;

    for (int e = wid; e < EE; e += nw) {
        const int s = ei[e];
        const int d = ei[EE + e];

        // prefetch next edge's M1 row (lowers to global_prefetch_b8)
        if (e + nw < EE) {
            const int sn = ei[e + nw];
            __builtin_prefetch(M1 + (size_t)sn * MSZ + lane * 16, 0, 0);
        }

        const float* ms = M1    + (size_t)s * MSZ;
        const float* qd = Q     + (size_t)d * HID;
        const float* k1 = Kagg1 + (size_t)s * HID;

        atomicAdd(Kagg2 + (size_t)d * HID + lane,      k1[lane]);
        atomicAdd(Kagg2 + (size_t)d * HID + 32 + lane, k1[32 + lane]);

        #pragma unroll
        for (int pass = 0; pass < 2; ++pass) {
            const int c = lane + 32 * pass;
            if (c < HV) {
                const int h = c / 10;
                const int j = c - h * 10;
                float acc = 0.f;
                #pragma unroll
                for (int i = 0; i < 16; ++i)
                    acc += qd[h * 16 + i] * ms[(h * 16 + i) * 10 + j];
                atomicAdd(H + (size_t)d * HV + c, acc);
            }
        }
    }
}

// =====================================================================
// Kernel 4: per-node combine for one hop:
//  Cd[h] = Q[n,h].Kagg[n,h] + CST
//  hidden[n,h,j] += hopwise[hop+1]*softmax(headwise[:,hop])[h] * H[n,h,j]/Cd[h]
// =====================================================================
__global__ __launch_bounds__(256) void k_combine(
    const float* __restrict__ Q, const float* __restrict__ Kagg,
    const float* __restrict__ H,
    const float* __restrict__ hopwise, const float* __restrict__ headwise,
    float* __restrict__ hidden, int hop)
{
    const int n = blockIdx.x * blockDim.x + threadIdx.x;
    if (n >= NN) return;

    float w[HEADS];
    float mx = -3.0e38f;
    #pragma unroll
    for (int h = 0; h < HEADS; ++h) {
        w[h] = headwise[h * 2 + hop];
        mx = w[h] > mx ? w[h] : mx;
    }
    float ssum = 0.f;
    #pragma unroll
    for (int h = 0; h < HEADS; ++h) { w[h] = __expf(w[h] - mx); ssum += w[h]; }
    const float g = hopwise[hop + 1];

    const float* q  = Q    + (size_t)n * HID;
    const float* ka = Kagg + (size_t)n * HID;
    const float* hh = H    + (size_t)n * HV;
    float*       hd = hidden + (size_t)n * HV;

    #pragma unroll
    for (int h = 0; h < HEADS; ++h) {
        float cd = CSTF;
        #pragma unroll
        for (int i = 0; i < HEADC; ++i) cd += q[h * 16 + i] * ka[h * 16 + i];
        const float scale = g * (w[h] / ssum) / cd;
        #pragma unroll
        for (int j = 0; j < NC; ++j) hd[h * 10 + j] += hh[h * 10 + j] * scale;
    }
}

// =====================================================================
// Kernel 5: out = hidden.reshape(N,40) @ W_out^T + b_out
// =====================================================================
__global__ __launch_bounds__(256) void k_out(
    const float* __restrict__ hidden,
    const float* __restrict__ Wout, const float* __restrict__ bout,
    float* __restrict__ out)
{
    const int n = blockIdx.x * blockDim.x + threadIdx.x;
    if (n >= NN) return;
    const float* hd = hidden + (size_t)n * HV;
    float hv[HV];
    #pragma unroll
    for (int k = 0; k < HV; ++k) hv[k] = hd[k];
    #pragma unroll
    for (int c = 0; c < NC; ++c) {
        float s = bout[c];
        #pragma unroll
        for (int k = 0; k < HV; ++k) s += hv[k] * Wout[c * HV + k];
        out[(size_t)n * NC + c] = s;
    }
}

// =====================================================================
extern "C" void kernel_launch(void* const* d_in, const int* in_sizes, int n_in,
                              void* d_out, int out_size, void* d_ws, size_t ws_size,
                              hipStream_t stream) {
    const float* feat    = (const float*)d_in[0];
    const int*   ei      = (const int*)  d_in[1];
    const float* Win     = (const float*)d_in[2];
    const float* bin     = (const float*)d_in[3];
    const float* Wq      = (const float*)d_in[4];
    const float* bq      = (const float*)d_in[5];
    const float* Wk      = (const float*)d_in[6];
    const float* bk      = (const float*)d_in[7];
    const float* Wv      = (const float*)d_in[8];
    const float* bv      = (const float*)d_in[9];
    const float* Wout    = (const float*)d_in[10];
    const float* bout    = (const float*)d_in[11];
    const float* hopwise = (const float*)d_in[12];
    const float* headwise= (const float*)d_in[13];
    float* out = (float*)d_out;

    // workspace layout (floats): ~190 MB total
    float* ws = (float*)d_ws;
    size_t o = 0;
    float* Q      = ws + o; o += (size_t)NN * HID;   // 64/node
    float* K0     = ws + o; o += (size_t)NN * HID;   // reused as Kagg2 in hop 2
    float* V0     = ws + o; o += (size_t)NN * HV;
    float* Kagg1  = ws + o; o += (size_t)NN * HID;
    float* H      = ws + o; o += (size_t)NN * HV;
    float* hidden = ws + o; o += (size_t)NN * HV;
    float* M1     = ws + o; o += (size_t)NN * MSZ;   // 128 MB, L2-resident

    hipMemsetAsync(Kagg1, 0, (size_t)NN * HID * sizeof(float), stream);
    hipMemsetAsync(H,     0, (size_t)NN * HV  * sizeof(float), stream);
    hipMemsetAsync(M1,    0, (size_t)NN * MSZ * sizeof(float), stream);

    // phase A: fused WMMA node transform (writes Q, K0, V0, hidden)
    k_node_transform<<<(NN + 127) / 128, 256, 0, stream>>>(
        feat, Win, bin, Wq, bq, Wk, bk, Wv, bv, hopwise, Q, K0, V0, hidden);

    // hop 1
    k_edge_hop1<<<1024, 256, 0, stream>>>(ei, Q, K0, V0, M1, Kagg1, H);
    k_combine<<<(NN + 255) / 256, 256, 0, stream>>>(
        Q, Kagg1, H, hopwise, headwise, hidden, 0);

    // hop 2 (K0 buffer recycled as Kagg2)
    hipMemsetAsync(H,  0, (size_t)NN * HV  * sizeof(float), stream);
    hipMemsetAsync(K0, 0, (size_t)NN * HID * sizeof(float), stream);
    k_edge_hop2<<<1024, 256, 0, stream>>>(ei, Q, Kagg1, M1, K0, H);
    k_combine<<<(NN + 255) / 256, 256, 0, stream>>>(
        Q, K0, H, hopwise, headwise, hidden, 1);

    // output projection
    k_out<<<(NN + 255) / 256, 256, 0, stream>>>(hidden, Wout, bout, out);
}
